// HarmonicSynthesizer_41420664603170
// MI455X (gfx1250) — matrix operand order repaired and checked by reading, not compiled, not run on hardware
//
#include <hip/hip_runtime.h>
#include <math.h>

#define T_STEPS   400
#define L_SAMP    96000
#define NH        100
#define NBATCH    2
#define SR_F      24000.0f
#define TWO_PI_F  6.2831853071795864f
#define NK        16001     // kept low-pass bins: k = 0..16000  (k*0.25Hz <= 4000Hz)
#define CHUNK     1024
#define NCHUNK    94        // ceil(96000/1024)
#define NTILES_N  25        // 400/16

typedef __attribute__((ext_vector_type(16))) _Float16 v16h;
typedef __attribute__((ext_vector_type(8)))  float    v8f;

// ---------------------------------------------------------------------------
// linear-upsample index helper (matches reference upsample_linear)
// ---------------------------------------------------------------------------
__device__ __forceinline__ void up_ix(int l, int& i0, int& i1, float& w) {
    float pos = ((float)l + 0.5f) * (400.0f / 96000.0f) - 0.5f;
    pos = fminf(fmaxf(pos, 0.0f), 399.0f);
    float fi = floorf(pos);
    i0 = (int)fi;
    i1 = (i0 + 1 < T_STEPS) ? (i0 + 1) : (T_STEPS - 1);
    w  = pos - fi;
}

// ---------------------------------------------------------------------------
// Depthwise k=3 conv (cross-correlation, pad=1), no activation
// ---------------------------------------------------------------------------
__global__ __launch_bounds__(256)
void dwconv3(const float* __restrict__ X, const float* __restrict__ W,
             const float* __restrict__ Bb, float* __restrict__ Y, int C) {
    int idx = blockIdx.x * blockDim.x + threadIdx.x;
    int total = NBATCH * C * T_STEPS;
    if (idx >= total) return;
    int t = idx % T_STEPS;
    int c = (idx / T_STEPS) % C;
    const float* x = X + (size_t)(idx - t);  // base of this (b,c) row
    float xm = (t > 0)            ? x[t - 1] : 0.0f;
    float xc = x[t];
    float xp = (t < T_STEPS - 1)  ? x[t + 1] : 0.0f;
    Y[idx] = W[c * 3 + 0] * xm + W[c * 3 + 1] * xc + W[c * 3 + 2] * xp + Bb[c];
}

// ---------------------------------------------------------------------------
// Fragment packing into exact CDNA5 WMMA VGPR order (ISA §7.12.2).
//  A (16-bit, 16x32 MxK): lane half = lane>>4, row = m0 + (lane&15),
//      slot j2 -> K = 16*(j2>>3) + 8*half + (j2&7)
//  B (16-bit, 32x16 KxN): col = n0 + (lane&15), slot j2 -> K = 16*(j2>>3) + 8*half + (j2&7)
// Layout in memory: frag = [tile...][lane] of 16 f16 (32B, aligned).
// ---------------------------------------------------------------------------
template <int M, int K>
__global__ __launch_bounds__(256)
void packA(const float* __restrict__ W, _Float16* __restrict__ apack) {
    constexpr int Kt = K / 32;
    constexpr int Mt = (M + 15) / 16;
    int idx = blockIdx.x * blockDim.x + threadIdx.x;   // (mt*Kt+kt)*32 + lane
    if (idx >= Mt * Kt * 32) return;
    int lane = idx & 31;
    int frag = idx >> 5;
    int kt = frag % Kt;
    int mt = frag / Kt;
    int half = lane >> 4, mr = lane & 15;
    int row = mt * 16 + mr;
    v16h f;
    #pragma unroll
    for (int j2 = 0; j2 < 16; ++j2) {
        int k = kt * 32 + ((j2 >> 3) << 4) + (half << 3) + (j2 & 7);
        float v = (row < M) ? W[row * K + k] : 0.0f;
        f[j2] = (_Float16)v;
    }
    *(v16h*)(apack + (size_t)idx * 16) = f;
}

// MODE 0: Bmat[k,t] = X[b,k,t]  (K == CIN)
// MODE 1: k = ic*3+dt, Bmat[k,t] = X[b,ic,t+dt-1] zero-padded  (K == 3*CIN)
template <int K, int CIN, int MODE>
__global__ __launch_bounds__(256)
void packB(const float* __restrict__ X, _Float16* __restrict__ bpack) {
    constexpr int Kt = K / 32;
    int idx = blockIdx.x * blockDim.x + threadIdx.x;   // ((b*Kt+kt)*25+nt)*32 + lane
    if (idx >= NBATCH * Kt * NTILES_N * 32) return;
    int lane = idx & 31;
    int rest = idx >> 5;
    int nt = rest % NTILES_N;
    int kt = (rest / NTILES_N) % Kt;
    int b  = rest / (NTILES_N * Kt);
    int half = lane >> 4;
    int col = nt * 16 + (lane & 15);
    const float* Xb = X + (size_t)b * CIN * T_STEPS;
    v16h f;
    #pragma unroll
    for (int j2 = 0; j2 < 16; ++j2) {
        int k = kt * 32 + ((j2 >> 3) << 4) + (half << 3) + (j2 & 7);
        float v;
        if (MODE == 0) {
            v = Xb[k * T_STEPS + col];
        } else {
            int ic = k / 3;
            int dt = k - ic * 3;
            int t  = col + dt - 1;
            v = (t >= 0 && t < T_STEPS) ? Xb[ic * T_STEPS + t] : 0.0f;
        }
        f[j2] = (_Float16)v;
    }
    *(v16h*)(bpack + (size_t)idx * 16) = f;
}

// ---------------------------------------------------------------------------
// WMMA GEMM on packed fragments: Y[b,m,t] = act( A*B + bias[m] )
//   ACT: 0 none, 1 leaky_relu(0.1), 2 softplus, 3 sigmoid
// One wave per 16x16 tile; inner loop = 2x v16h load + v_wmma, no branches.
// For M % 16 == 0 the store epilogue is fully unconditional (no exec writes).
// ---------------------------------------------------------------------------
template <int M, int K, int ACT>
__global__ __launch_bounds__(32)
void gemm16(const _Float16* __restrict__ apack, const _Float16* __restrict__ bpack,
            const float* __restrict__ bias, float* __restrict__ Y) {
    constexpr int Kt = K / 32;
    const int lane = threadIdx.x;
    const int half = lane >> 4;
    const int mt = blockIdx.x;
    const int nt = blockIdx.y;
    const int b  = blockIdx.z;
    const _Float16* ap = apack + ((size_t)(mt * Kt) * 32 + lane) * 16;
    const _Float16* bp = bpack + (((size_t)(b * Kt) * NTILES_N + nt) * 32 + lane) * 16;

    v8f acc = {};
    #pragma unroll
    for (int kt = 0; kt < Kt; ++kt) {
        v16h afrag = *(const v16h*)(ap + (size_t)kt * 32 * 16);
        v16h bfrag = *(const v16h*)(bp + (size_t)kt * NTILES_N * 32 * 16);
        acc = __builtin_amdgcn_wmma_f32_16x16x32_f16(false, afrag, false, bfrag,
                                                     (short)0, acc, false, false);
    }
    const int col = nt * 16 + (lane & 15);
    #pragma unroll
    for (int v = 0; v < 8; ++v) {
        int row = mt * 16 + v + half * 8;
        bool ok = (M % 16 == 0) ? true : (row < M);   // compile-time elision when M%16==0
        if (ok) {
            float val = acc[v] + bias[row];
            if (ACT == 1)      val = (val > 0.0f) ? val : 0.1f * val;
            else if (ACT == 2) val = fmaxf(val, 0.0f) + log1pf(expf(-fabsf(val)));
            else if (ACT == 3) val = 1.0f / (1.0f + expf(-val));
            Y[((size_t)b * M + row) * T_STEPS + col] = val;
        }
    }
}

// ---------------------------------------------------------------------------
// Harmonic shaping (in place, at T resolution — matches reference order)
// ---------------------------------------------------------------------------
__global__ __launch_bounds__(256)
void shape_kernel(float* __restrict__ ha, float* __restrict__ hfn) {
    int idx = blockIdx.x * blockDim.x + threadIdx.x;
    if (idx >= NBATCH * NH * T_STEPS) return;
    int h = (idx / T_STEPS) % NH;
    float tilt = (float)h / 99.0f;                       // (h_idx-1)/(H-1)
    float a = ha[idx];
    float s = fmaxf(a, a * (0.15f * tilt));
    float z = ((float)h - 66.0f) / (100.0f / 6.0f);      // upper_mid = 2H//3 = 66
    s *= 1.0f + expf(-0.5f * z * z);
    ha[idx] = s;
    float f = hfn[idx] + 0.5f * tilt;
    hfn[idx] = fminf(fmaxf(f, 0.0f), 1.0f);
}

// ---------------------------------------------------------------------------
// Phase = cumsum(2*pi*f0_up/SR): 3-kernel scan over L per batch
// ---------------------------------------------------------------------------
__global__ __launch_bounds__(1024)
void phaseA(const float* __restrict__ f0, float* __restrict__ f0_up,
            float* __restrict__ csum) {
    __shared__ float sm[1024];
    int cid = blockIdx.x, b = blockIdx.y, tid = threadIdx.x;
    int l = cid * CHUNK + tid;
    float om = 0.0f;
    if (l < L_SAMP) {
        int i0, i1; float w;
        up_ix(l, i0, i1, w);
        float f = f0[b * T_STEPS + i0] * (1.0f - w) + f0[b * T_STEPS + i1] * w;
        f0_up[(size_t)b * L_SAMP + l] = f;
        om = TWO_PI_F * f / SR_F;
    }
    sm[tid] = om; __syncthreads();
    for (int off = 512; off > 0; off >>= 1) {
        if (tid < off) sm[tid] += sm[tid + off];
        __syncthreads();
    }
    if (tid == 0) csum[b * NCHUNK + cid] = sm[0];
}

__global__ __launch_bounds__(64)
void phaseB(float* __restrict__ csum) {
    int tid = threadIdx.x;
    if (tid < NBATCH) {
        float run = 0.0f;
        for (int i = 0; i < NCHUNK; ++i) {
            float s = csum[tid * NCHUNK + i];
            csum[tid * NCHUNK + i] = run;   // exclusive offsets
            run += s;
        }
    }
}

__global__ __launch_bounds__(1024)
void phaseC(const float* __restrict__ f0_up, const float* __restrict__ csum,
            float* __restrict__ phase) {
    __shared__ float sm[1024];
    int cid = blockIdx.x, b = blockIdx.y, tid = threadIdx.x;
    int l = cid * CHUNK + tid;
    float om = 0.0f;
    if (l < L_SAMP) om = TWO_PI_F * f0_up[(size_t)b * L_SAMP + l] / SR_F;
    sm[tid] = om; __syncthreads();
    for (int off = 1; off < 1024; off <<= 1) {    // Hillis-Steele inclusive scan
        float add = (tid >= off) ? sm[tid - off] : 0.0f;
        __syncthreads();
        sm[tid] += add;
        __syncthreads();
    }
    if (l < L_SAMP) phase[(size_t)b * L_SAMP + l] = sm[tid] + csum[b * NCHUNK + cid];
}

// ---------------------------------------------------------------------------
// Main per-sample synthesis: sum over H=100 harmonics.
// A 256-thread block spans <= 4 condition-time columns -> stage the
// shaped/hfn interpolation columns in LDS once per block.
// ---------------------------------------------------------------------------
__global__ __launch_bounds__(256)
void synth(const float* __restrict__ f0_up, const float* __restrict__ phase,
           const float* __restrict__ shaped, const float* __restrict__ hfn,
           const float* __restrict__ bf, const float* __restrict__ jitter,
           const float* __restrict__ shimmer, const float* __restrict__ breathiness,
           const float* __restrict__ breath_signal,
           const float* __restrict__ phase_noise, const float* __restrict__ harm_noise,
           const float* __restrict__ jit_noise, float* __restrict__ out) {
    __shared__ float s_sh[NH * 4];
    __shared__ float s_hf[NH * 4];
    int l = blockIdx.x * blockDim.x + threadIdx.x;
    int b = blockIdx.y;

    int t0, tu; float wu;
    up_ix(blockIdx.x * 256, t0, tu, wu);    // block-base column
    const size_t bT = (size_t)b * NH * T_STEPS;
    for (int i = threadIdx.x; i < NH * 4; i += 256) {
        int h = i >> 2, dt = i & 3;
        int t = t0 + dt; if (t > T_STEPS - 1) t = T_STEPS - 1;
        s_sh[i] = shaped[bT + h * T_STEPS + t];
        s_hf[i] = hfn[bT + h * T_STEPS + t];
    }
    __syncthreads();

    int i0, i1; float w;
    up_ix(l, i0, i1, w);
    const int d0 = i0 - t0, d1 = i1 - t0;   // 0..3 within staged columns
    const size_t bl = (size_t)b * L_SAMP + l;

    float f0u = f0_up[bl];
    float inh = bf[b * 1200 + 400 + i0] * (1.0f - w) + bf[b * 1200 + 400 + i1] * w;
    float exh = bf[b * 1200 + 800 + i0] * (1.0f - w) + bf[b * 1200 + 800 + i1] * w;
    float jp  = phase[bl] + jitter[bl] * (1.0f + inh * 0.5f) * jit_noise[bl];

    const float* pn = phase_noise + (size_t)b * NH * L_SAMP + l;
    const float* hm = harm_noise  + (size_t)b * NH * L_SAMP + l;

    float acc = 0.0f;
    #pragma unroll 4
    for (int h = 0; h < NH; ++h) {
        float hn1  = (float)(h + 1);
        float mask = (hn1 * f0u < 0.5f * SR_F) ? 1.0f : 0.0f;
        float amp  = s_sh[(h << 2) + d0] * (1.0f - w) + s_sh[(h << 2) + d1] * w;
        float hfu  = s_hf[(h << 2) + d0] * (1.0f - w) + s_hf[(h << 2) + d1] * w;
        float disp = (hn1 > 20.0f) ? (hn1 - 20.0f) * (1.0f / 80.0f) : 0.0f;
        float phh  = jp * hn1 + pn[(size_t)h * L_SAMP] * 0.2f * disp;
        float s    = sinf(phh);
        float wav  = (1.0f - hfu) * s + hfu * hm[(size_t)h * L_SAMP];
        acc += amp * wav * mask;
    }
    float shim = shimmer[bl] * (1.0f + exh * 0.3f);
    float enh  = acc * (1.0f + shim) * (1.0f - 0.3f * breathiness[bl]);
    float voiced = (f0u > 0.0f) ? 1.0f : 0.0f;
    out[bl] = enh * voiced + breath_signal[bl] * (1.0f - voiced);
}

// ---------------------------------------------------------------------------
// Brick-wall high-pass via low-pass DFT subtraction.
// Incremental Givens rotation, exact (k*n mod L) resync every 256 steps.
// ---------------------------------------------------------------------------
__global__ __launch_bounds__(256)
void dft_fwd(const float* __restrict__ x, float* __restrict__ Xre,
             float* __restrict__ Xim) {
    int k = blockIdx.x * blockDim.x + threadIdx.x;
    int b = blockIdx.y;
    if (k >= NK) return;
    const float* xb = x + (size_t)b * L_SAMP;
    const double w0 = 6.283185307179586476 / (double)L_SAMP;
    float ca = (float)cos(w0 * (double)k), sa = (float)sin(w0 * (double)k);
    float ar = 0.0f, ai = 0.0f;
    for (int n0 = 0; n0 < L_SAMP; n0 += 256) {
        long long m = ((long long)k * (long long)n0) % L_SAMP;
        float c = (float)cos(w0 * (double)m);
        float s = (float)sin(w0 * (double)m);
        for (int t = 0; t < 256; ++t) {
            float xv = xb[n0 + t];             // shared n across lanes -> broadcast
            ar = fmaf(xv, c, ar);
            ai = fmaf(-xv, s, ai);
            float cn = c * ca - s * sa;
            s = fmaf(s, ca, c * sa);
            c = cn;
        }
    }
    Xre[(size_t)b * NK + k] = ar;
    Xim[(size_t)b * NK + k] = ai;
}

__global__ __launch_bounds__(256)
void dft_inv_add(const float* __restrict__ x, const float* __restrict__ Xre,
                 const float* __restrict__ Xim, const float* __restrict__ f0_up,
                 float* __restrict__ out) {
    int n = blockIdx.x * blockDim.x + threadIdx.x;
    int b = blockIdx.y;
    const float* Re = Xre + (size_t)b * NK;
    const float* Im = Xim + (size_t)b * NK;
    const double w0 = 6.283185307179586476 / (double)L_SAMP;
    float ca = (float)cos(w0 * (double)n), sa = (float)sin(w0 * (double)n);
    float S = 0.0f;
    for (int k0 = 0; k0 < NK; k0 += 256) {
        long long m = ((long long)n * (long long)k0) % L_SAMP;
        float c = (float)cos(w0 * (double)m);
        float s = (float)sin(w0 * (double)m);
        int kmax = (NK - k0 < 256) ? (NK - k0) : 256;
        for (int t = 0; t < kmax; ++t) {
            S += Re[k0 + t] * c - Im[k0 + t] * s;   // e^{+i theta} real part
            float cn = c * ca - s * sa;
            s = fmaf(s, ca, c * sa);
            c = cn;
        }
    }
    float lp  = (2.0f * S - Re[0]) * (1.0f / (float)L_SAMP);
    size_t bl = (size_t)b * L_SAMP + n;
    float air = 0.05f * (x[bl] - lp);
    float voiced = (f0_up[bl] > 0.0f) ? 1.0f : 0.0f;
    out[bl] += air * ((1.0f - voiced) * 0.3f + 0.05f);
}

// ---------------------------------------------------------------------------
extern "C" void kernel_launch(void* const* d_in, const int* in_sizes, int n_in,
                              void* d_out, int out_size, void* d_ws, size_t ws_size,
                              hipStream_t stream) {
    (void)in_sizes; (void)n_in; (void)out_size; (void)ws_size;
    const float* f0            = (const float*)d_in[0];
    const float* condition     = (const float*)d_in[1];
    const float* dw1_w = (const float*)d_in[2];
    const float* dw1_b = (const float*)d_in[3];
    const float* pw1_w = (const float*)d_in[4];
    const float* pw1_b = (const float*)d_in[5];
    const float* dw2_w = (const float*)d_in[6];
    const float* dw2_b = (const float*)d_in[7];
    const float* pw2_w = (const float*)d_in[8];
    const float* pw2_b = (const float*)d_in[9];
    const float* pw3_w = (const float*)d_in[10];
    const float* pw3_b = (const float*)d_in[11];
    const float* hf1_w = (const float*)d_in[12];
    const float* hf1_b = (const float*)d_in[13];
    const float* hf2_w = (const float*)d_in[14];
    const float* hf2_b = (const float*)d_in[15];
    const float* hf3_w = (const float*)d_in[16];
    const float* hf3_b = (const float*)d_in[17];
    const float* jitter        = (const float*)d_in[18];
    const float* shimmer       = (const float*)d_in[19];
    const float* breathiness   = (const float*)d_in[20];
    const float* breath_signal = (const float*)d_in[21];
    const float* bf            = (const float*)d_in[22];
    const float* phase_noise   = (const float*)d_in[23];
    const float* harm_noise    = (const float*)d_in[24];
    const float* jit_noise     = (const float*)d_in[25];
    const float* air_noise     = (const float*)d_in[26];
    float* out = (float*)d_out;

    // workspace layout (float offsets)
    float* ws     = (float*)d_ws;
    float* x1     = ws;                    // 2*128*400 = 102400
    float* x2     = x1 + 102400;           // 2*256*400 = 204800
    float* x3     = x2 + 204800;
    float* x4     = x3 + 204800;
    float* shaped = x4 + 204800;           // 2*100*400 = 80000
    float* y1     = shaped + 80000;        // 102400
    float* y2     = y1 + 102400;           // 2*64*400 = 51200
    float* hfnb   = y2 + 51200;            // 80000
    float* f0_up  = hfnb + 80000;          // 192000
    float* phase  = f0_up + 192000;        // 192000
    float* csum   = phase + 192000;        // 256
    float* Xre    = csum + 256;            // 32002
    float* Xim    = Xre + 32002;           // 32002 -> ends 1478660
    _Float16* apack = (_Float16*)(ws + 1478672);           // max 65536 halves (32B aligned)
    _Float16* bpack = (_Float16*)(ws + 1478672 + 32768);   // max 307200 halves

    #define PKA(Mv,Kv,Wp)        packA<Mv,Kv><<<dim3(((Mv+15)/16*(Kv/32)*32 + 255)/256), 256, 0, stream>>>(Wp, apack)
    #define PKB(Kv,Cv,MODE,Xp)   packB<Kv,Cv,MODE><<<dim3((NBATCH*(Kv/32)*NTILES_N*32 + 255)/256), 256, 0, stream>>>(Xp, bpack)

    // conv front-end (WMMA for all dense GEMMs, fragments pre-packed)
    dwconv3<<<dim3((NBATCH*128*T_STEPS + 255)/256), 256, 0, stream>>>(condition, dw1_w, dw1_b, x1, 128);
    PKA(256, 128, pw1_w);  PKB(128, 128, 0, x1);
    gemm16<256, 128, 1><<<dim3(16, NTILES_N, NBATCH), 32, 0, stream>>>(apack, bpack, pw1_b, x2);
    dwconv3<<<dim3((NBATCH*256*T_STEPS + 255)/256), 256, 0, stream>>>(x2, dw2_w, dw2_b, x3, 256);
    PKA(256, 256, pw2_w);  PKB(256, 256, 0, x3);
    gemm16<256, 256, 1><<<dim3(16, NTILES_N, NBATCH), 32, 0, stream>>>(apack, bpack, pw2_b, x4);
    PKA(100, 256, pw3_w);  PKB(256, 256, 0, x4);
    gemm16<100, 256, 2><<<dim3(7, NTILES_N, NBATCH), 32, 0, stream>>>(apack, bpack, pw3_b, shaped);
    PKA(128, 384, hf1_w);  PKB(384, 128, 1, condition);
    gemm16<128, 384, 1><<<dim3(8, NTILES_N, NBATCH), 32, 0, stream>>>(apack, bpack, hf1_b, y1);
    PKA(64, 384, hf2_w);   PKB(384, 128, 1, y1);
    gemm16<64, 384, 1><<<dim3(4, NTILES_N, NBATCH), 32, 0, stream>>>(apack, bpack, hf2_b, y2);
    PKA(100, 64, hf3_w);   PKB(64, 64, 0, y2);
    gemm16<100, 64, 3><<<dim3(7, NTILES_N, NBATCH), 32, 0, stream>>>(apack, bpack, hf3_b, hfnb);
    shape_kernel<<<dim3((NBATCH*NH*T_STEPS + 255)/256), 256, 0, stream>>>(shaped, hfnb);

    // phase cumsum
    phaseA<<<dim3(NCHUNK, NBATCH), 1024, 0, stream>>>(f0, f0_up, csum);
    phaseB<<<1, 64, 0, stream>>>(csum);
    phaseC<<<dim3(NCHUNK, NBATCH), 1024, 0, stream>>>(f0_up, csum, phase);

    // harmonic synthesis (dominant: 153.6 MB noise reads, 19.2M sinf)
    synth<<<dim3(L_SAMP/256, NBATCH), 256, 0, stream>>>(f0_up, phase, shaped, hfnb, bf,
        jitter, shimmer, breathiness, breath_signal, phase_noise, harm_noise, jit_noise, out);

    // high-pass air via low-pass DFT subtraction
    dft_fwd<<<dim3((NK + 255)/256, NBATCH), 256, 0, stream>>>(air_noise, Xre, Xim);
    dft_inv_add<<<dim3(L_SAMP/256, NBATCH), 256, 0, stream>>>(air_noise, Xre, Xim, f0_up, out);

    #undef PKA
    #undef PKB
}
